// Attention_27487790694697
// MI455X (gfx1250) — matrix operand order repaired and checked by reading, not compiled, not run on hardware
//
#include <hip/hip_runtime.h>
#include <hip/hip_bf16.h>

typedef __bf16 bf16;
typedef __attribute__((ext_vector_type(16))) __bf16 v16bf;
typedef __attribute__((ext_vector_type(8)))  float  v8f;
typedef __attribute__((ext_vector_type(4)))  int    v4i;

#define DEVINL __device__ __forceinline__

// ---------------- fixed problem shape (from reference setup_inputs) ----------
constexpr int Ntok = 4096;   // tokens per branch
constexpr int C    = 512;    // channels
constexpr int Wimg = 64;     // H = W = 64
constexpr int Mred = 1024;   // tokens after 2x2 spatial reduction
constexpr int HEADS = 8, HD = 64;
constexpr float LNEPS  = 1e-5f;
constexpr float THRESH = 0.02f;
constexpr float QSCALE = 0.125f;  // hd^-0.5, exact power of two -> fold into Q

// ---------------- GEMM tiling ----------------
constexpr int BM = 128, BN = 128, BK = 32;
constexpr int LAS = BK + 8;    // LDS row stride (bf16): 40 elems = 80 B (16B-mult)
constexpr int LBK = BK + 8;

// ---------------- async copy to LDS (gfx1250 path, guarded) ----------------
#if __has_builtin(__builtin_amdgcn_global_load_async_to_lds_b128) && \
    __has_builtin(__builtin_amdgcn_s_wait_asynccnt)
#define ASYNC_OK 1
#define AS1 __attribute__((address_space(1)))
#define AS3 __attribute__((address_space(3)))
DEVINL void cp16(const bf16* g, bf16* l) {  // 16 bytes, async engine
  __builtin_amdgcn_global_load_async_to_lds_b128((AS1 v4i*)g, (AS3 v4i*)l,
                                                 0, 0);
}
DEVINL void cp_commit() { __builtin_amdgcn_s_wait_asynccnt(0); }
#else
DEVINL void cp16(const bf16* g, bf16* l) {
  *(float4*)l = *(const float4*)g;
}
DEVINL void cp_commit() {}
#endif

// ---------------- WMMA helper ----------------
DEVINL v8f wmma_bf16(v16bf a, v16bf b, v8f c) {
  return __builtin_amdgcn_wmma_f32_16x16x32_bf16(
      false, a, false, b, (short)0, c, false, false);
}

// A-matrix 16x32 bf16 fragment: element (m,k) at p[m*rs + k] (k contiguous).
DEVINL v16bf frag_a(const bf16* p, int rs) {
  const int lane = threadIdx.x & 31;
  const bf16* q = p + (lane & 15) * rs + ((lane & 16) ? 8 : 0);
  v16bf f;
#pragma unroll
  for (int v = 0; v < 8; ++v) {
    const int k = ((v & 4) ? 16 : 0) + 2 * (v & 3);
    f[2 * v]     = q[k];
    f[2 * v + 1] = q[k + 1];
  }
  return f;
}

// B-matrix 32x16 bf16 fragment from a transposed tile: element (k,n) at
// p[n*cs + k] (k contiguous -> two ds_load_b128 per fragment).
DEVINL v16bf frag_bt(const bf16* p, int cs) {
  const int lane = threadIdx.x & 31;
  const bf16* q = p + (lane & 15) * cs + ((lane & 16) ? 8 : 0);
  v16bf f;
#pragma unroll
  for (int v = 0; v < 8; ++v) {
    const int k = ((v & 4) ? 16 : 0) + 2 * (v & 3);
    f[2 * v]     = q[k];
    f[2 * v + 1] = q[k + 1];
  }
  return f;
}

// ---- 16-lane-row reductions via DS_SWIZZLE_B32 ----
template <int PAT>
DEVINL float swz(float v) {
  return __int_as_float(__builtin_amdgcn_ds_swizzle(__float_as_int(v), PAT));
}
DEVINL float rowmax16(float x) {
  x = fmaxf(x, swz<0x041f>(x));
  x = fmaxf(x, swz<0x081f>(x));
  x = fmaxf(x, swz<0x101f>(x));
  x = fmaxf(x, swz<0x201f>(x));
  return x;
}
DEVINL float rowsum16(float x) {
  x += swz<0x041f>(x);
  x += swz<0x081f>(x);
  x += swz<0x101f>(x);
  x += swz<0x201f>(x);
  return x;
}

// ---------------- precision/layout prep kernels ----------------
__global__ void __launch_bounds__(256)
convert_bf16(const float* __restrict__ in, bf16* __restrict__ out) {
  const int i = (blockIdx.x * 256 + threadIdx.x) * 8;
  const float4 a = *(const float4*)(in + i);
  const float4 b = *(const float4*)(in + i + 4);
  out[i + 0] = (bf16)a.x; out[i + 1] = (bf16)a.y;
  out[i + 2] = (bf16)a.z; out[i + 3] = (bf16)a.w;
  out[i + 4] = (bf16)b.x; out[i + 5] = (bf16)b.y;
  out[i + 6] = (bf16)b.z; out[i + 7] = (bf16)b.w;
}

// W [K,N] f32 (row-major) -> Wt [N,K] bf16
__global__ void __launch_bounds__(256)
transpose_convert(const float* __restrict__ W, bf16* __restrict__ Wt,
                  int K, int N) {
  __shared__ bf16 t[64][65];
  const int bn = blockIdx.x * 64, bk = blockIdx.y * 64;
  const int tid = threadIdx.x;
#pragma unroll
  for (int j = 0; j < 16; ++j) {
    const int idx = tid + j * 256;
    const int r = idx >> 6, c = idx & 63;  // r = k-local, c = n-local
    t[c][r] = (bf16)W[(size_t)(bk + r) * N + bn + c];
  }
  __syncthreads();
#pragma unroll
  for (int j = 0; j < 16; ++j) {
    const int idx = tid + j * 256;
    const int r = idx >> 6, c = idx & 63;  // r = n-local, c = k-local
    Wt[(size_t)(bn + r) * K + bk + c] = t[r][c];
  }
}

// Wsr [o][i][dy][dx] f32 -> Wt[o][kk'] bf16 with kk' = (dy*2+dx)*512 + i
__global__ void __launch_bounds__(256)
convert_wsr(const float* __restrict__ Wsr, bf16* __restrict__ out) {
  const int idx = blockIdx.x * 256 + threadIdx.x;  // 512*2048 elements
  const int o = idx >> 11, kk = idx & 2047;
  const int pp = kk >> 9, i = kk & 511;
  out[idx] = (bf16)Wsr[(size_t)o * 2048 + i * 4 + pp];
}

// ---------------- generic WMMA GEMM, async double-buffered ----------------
// A is bf16. AMODE 0: A[m*lda + k].  AMODE 1: SR-conv remapped gather:
//   kk' = patch*512 + channel; tile rows are contiguous 32-elem runs of x_bf.
// B is pre-transposed bf16 [n][k], stride ldb.
// OUTMODE 0: f32 out, (acc + bias[n]) * rowmask[m]
// OUTMODE 1: bf16 out, acc * outscale
// OUTMODE 2: KV split: col<C -> K_bf[row*C+col]; else V_bf[(col-C)*Mred+row]
template <int AMODE, int OUTMODE>
__global__ void __launch_bounds__(256)
gemm_wmma(const bf16* __restrict__ A, const bf16* __restrict__ Bt,
          const float* __restrict__ bias, const float* __restrict__ rowmask,
          float* __restrict__ Cf, bf16* __restrict__ Cb0,
          bf16* __restrict__ Cb1, float outscale,
          int Kdim, int lda, int ldb, int ldc) {
  __shared__ bf16 As[2][BM][LAS];
  __shared__ bf16 Bs[2][BN][LBK];
  const int tid  = threadIdx.x;
  const int m0   = blockIdx.y * BM;
  const int n0   = blockIdx.x * BN;
  const int wave = tid >> 5;
  const int wm   = wave & 3;
  const int wn   = wave >> 2;
  const int lane = tid & 31;

  v8f acc[2][4] = {};

  auto stageA = [&](int kk, int buf) {
#pragma unroll
    for (int j = 0; j < 2; ++j) {            // 512 chunks of 16B
      const int ch = tid + j * 256;
      const int r = ch >> 2, q = (ch & 3) * 8;
      const bf16* src;
      if (AMODE == 0) {
        src = A + (size_t)(m0 + r) * lda + kk + q;
      } else {
        const int m  = m0 + r;
        const int pp = (kk + q) >> 9;        // patch index dy*2+dx
        const int ci = (kk + q) & 511;       // channel
        const int dy = pp >> 1, dx = pp & 1;
        const int my = m >> 5, mx = m & 31;
        src = A + ((size_t)((2 * my + dy) * Wimg + 2 * mx + dx)) * C + ci;
      }
      cp16(src, &As[buf][r][q]);
    }
  };
  auto stageB = [&](int kk, int buf) {
#pragma unroll
    for (int j = 0; j < 2; ++j) {
      const int ch = tid + j * 256;
      const int r = ch >> 2, q = (ch & 3) * 8;
      cp16(Bt + (size_t)(n0 + r) * ldb + kk + q, &Bs[buf][r][q]);
    }
  };
  auto prefetchAB = [&](int kk) {  // -> global_prefetch_b8
    const int r = tid >> 2, q = (tid & 3) * 8;
    if (AMODE == 0)
      __builtin_prefetch(A + (size_t)(m0 + r) * lda + kk + q, 0, 1);
    __builtin_prefetch(Bt + (size_t)(n0 + r) * ldb + kk + q, 0, 1);
  };

  stageA(0, 0); stageB(0, 0);
  cp_commit();
  __syncthreads();

  for (int k0 = 0;; k0 += BK) {
    const int  cur = (k0 / BK) & 1;
    const bool has_next = (k0 + BK < Kdim);
    if (has_next) { stageA(k0 + BK, cur ^ 1); stageB(k0 + BK, cur ^ 1); }
    if (k0 + 2 * BK < Kdim) prefetchAB(k0 + 2 * BK);

    const v16bf a0 = frag_a(&As[cur][wm * 32][0], LAS);
    const v16bf a1 = frag_a(&As[cur][wm * 32 + 16][0], LAS);
#pragma unroll
    for (int t = 0; t < 4; ++t) {
      const v16bf b = frag_bt(&Bs[cur][wn * 64 + t * 16][0], LBK);
      acc[0][t] = wmma_bf16(a0, b, acc[0][t]);
      acc[1][t] = wmma_bf16(a1, b, acc[1][t]);
    }
    if (!has_next) break;
    cp_commit();       // own async copies for next tile have landed
    __syncthreads();   // everyone done reading cur & copies visible
  }

  const int rb = (lane & 16) ? 8 : 0;
  const int cn = lane & 15;
#pragma unroll
  for (int i = 0; i < 2; ++i) {
#pragma unroll
    for (int t = 0; t < 4; ++t) {
      const int col = n0 + wn * 64 + t * 16 + cn;
      const float bv = (OUTMODE == 0 && bias) ? bias[col] : 0.f;
#pragma unroll
      for (int r = 0; r < 8; ++r) {
        const int row = m0 + wm * 32 + i * 16 + rb + r;
        float v = acc[i][t][r];
        if (OUTMODE == 0) {
          v += bv;
          if (rowmask) v *= rowmask[row];
          Cf[(size_t)row * ldc + col] = v;
        } else if (OUTMODE == 1) {
          Cb0[(size_t)row * ldc + col] = (bf16)(v * outscale);
        } else {
          if (col < C) Cb0[(size_t)row * C + col] = (bf16)v;
          else         Cb1[(size_t)(col - C) * Mred + row] = (bf16)v;
        }
      }
    }
  }
}

// ---------------- LayerNorm (f32 in, bf16 out), one wave per row ----------
__global__ void __launch_bounds__(256)
layernorm_rows(const float* __restrict__ x, bf16* __restrict__ y,
               const float* __restrict__ gamma, const float* __restrict__ beta,
               int rows) {
  const int wave = threadIdx.x >> 5, lane = threadIdx.x & 31;
  const int row  = blockIdx.x * 8 + wave;
  if (row >= rows) return;
  const float* p = x + (size_t)row * C;
  float v[16];
  float s = 0.f;
#pragma unroll
  for (int j = 0; j < 16; ++j) { v[j] = p[lane + j * 32]; s += v[j]; }
#pragma unroll
  for (int m = 1; m < 32; m <<= 1) s += __shfl_xor(s, m);
  const float mu = s * (1.0f / C);
  float q = 0.f;
#pragma unroll
  for (int j = 0; j < 16; ++j) { const float d = v[j] - mu; q += d * d; }
#pragma unroll
  for (int m = 1; m < 32; m <<= 1) q += __shfl_xor(q, m);
  const float inv = rsqrtf(q * (1.0f / C) + LNEPS);
#pragma unroll
  for (int j = 0; j < 16; ++j) {
    const int c = lane + j * 32;
    y[(size_t)row * C + c] = (bf16)((v[j] - mu) * inv * gamma[c] + beta[c]);
  }
}

// ---------------- flash-style attention (all-bf16 operands) ----------------
constexpr int BQ = 128, BKV = 64;
constexpr int LQ = HD + 8, LKV = HD + 8, LVT = BKV + 8, LP = BKV + 8;

__global__ void __launch_bounds__(256)
attn_wmma(const bf16* __restrict__ Q, const bf16* __restrict__ K,
          const bf16* __restrict__ V,  // V pre-transposed: [h*64+d][m]
          bf16* __restrict__ O) {
  __shared__ bf16 Qs[BQ][LQ];
  __shared__ bf16 Ks[BKV][LKV];
  __shared__ bf16 Vt[HD][LVT];
  __shared__ bf16 Ps[8][16][LP];
  const int tid  = threadIdx.x;
  const int h    = blockIdx.y;
  const int qb   = blockIdx.x;
  const int wave = tid >> 5, lane = tid & 31;
  const int rb   = (lane & 16) ? 8 : 0;
  const int cn   = lane & 15;

  // stage Q block once (Q already pre-scaled by 1/8 in its GEMM epilogue)
#pragma unroll
  for (int j = 0; j < 4; ++j) {
    const int ch = tid + j * 256;            // 1024 chunks of 16B
    const int r = ch >> 3, q = (ch & 7) * 8;
    cp16(Q + (size_t)(qb * BQ + r) * C + h * HD + q, &Qs[r][q]);
  }

  v8f oacc[4] = {};
  float mrun[8], lrun[8];
#pragma unroll
  for (int r = 0; r < 8; ++r) { mrun[r] = -3.0e38f; lrun[r] = 0.f; }

  constexpr int NKB = Mred / BKV;
  for (int kb = 0; kb < NKB; ++kb) {
    __syncthreads();
#pragma unroll
    for (int j = 0; j < 2; ++j) {            // 512 chunks each for K and Vt
      const int ch = tid + j * 256;
      const int r = ch >> 2, q = (ch & 3) * 8;
      cp16(K + (size_t)(kb * BKV + r) * C + h * HD + q, &Ks[r][q]);
      cp16(V + (size_t)(h * HD + r) * Mred + kb * BKV + q, &Vt[r][q]);
    }
    if (kb + 1 < NKB) {                      // L2 prefetch of next block
      const int r = tid >> 2, q = (tid & 3) * 8;
      __builtin_prefetch(K + (size_t)((kb + 1) * BKV + r) * C + h * HD + q,
                         0, 1);
      __builtin_prefetch(V + (size_t)(h * HD + r) * Mred + (kb + 1) * BKV + q,
                         0, 1);
    }
    cp_commit();
    __syncthreads();

    // S = Q * K^T
    const v16bf aq0 = frag_a(&Qs[wave * 16][0], LQ);
    const v16bf aq1 = frag_a(&Qs[wave * 16][32], LQ);
    v8f s[4];
#pragma unroll
    for (int t = 0; t < 4; ++t) {
      v8f z = {};
      z = wmma_bf16(aq0, frag_bt(&Ks[t * 16][0], LKV), z);
      z = wmma_bf16(aq1, frag_bt(&Ks[t * 16][32], LKV), z);
      s[t] = z;
    }

    // online softmax; each row lives on 16 lanes of one half-wave
#pragma unroll
    for (int r = 0; r < 8; ++r) {
      float mx = fmaxf(fmaxf(s[0][r], s[1][r]), fmaxf(s[2][r], s[3][r]));
      mx = rowmax16(mx);
      const float mn   = fmaxf(mrun[r], mx);
      const float corr = __expf(mrun[r] - mn);
      mrun[r] = mn;
      float ls = 0.f;
#pragma unroll
      for (int t = 0; t < 4; ++t) {
        const float pv = __expf(s[t][r] - mn);
        s[t][r] = pv;
        ls += pv;
      }
      ls = rowsum16(ls);
      lrun[r] = lrun[r] * corr + ls;
#pragma unroll
      for (int t = 0; t < 4; ++t) oacc[t][r] *= corr;
    }

    // stage P (bf16) to reload in A-fragment layout
#pragma unroll
    for (int t = 0; t < 4; ++t)
#pragma unroll
      for (int r = 0; r < 8; ++r)
        Ps[wave][rb + r][t * 16 + cn] = (bf16)s[t][r];
    __syncthreads();

    // O += P * V
    const v16bf ap0 = frag_a(&Ps[wave][0][0], LP);
    const v16bf ap1 = frag_a(&Ps[wave][0][32], LP);
#pragma unroll
    for (int t = 0; t < 4; ++t) {
      oacc[t] = wmma_bf16(ap0, frag_bt(&Vt[t * 16][0], LVT), oacc[t]);
      oacc[t] = wmma_bf16(ap1, frag_bt(&Vt[t * 16][32], LVT), oacc[t]);
    }
  }

#pragma unroll
  for (int t = 0; t < 4; ++t)
#pragma unroll
    for (int r = 0; r < 8; ++r) {
      const int row = qb * BQ + wave * 16 + rb + r;
      const int col = h * HD + t * 16 + cn;
      O[(size_t)row * C + col] = (bf16)(oacc[t][r] / lrun[r]);
    }
}

// ---------------- TokenExchange (float4-vectorized) ----------------
__global__ void __launch_bounds__(256)
exchange_kernel(const float* __restrict__ o0, const float* __restrict__ o1,
                const float* __restrict__ m0, const float* __restrict__ m1,
                float* __restrict__ out) {
  const int idx = (blockIdx.x * 256 + threadIdx.x) * 4;
  const int row = idx >> 9;
  const float4 a = *(const float4*)(o0 + idx);
  const float4 b = *(const float4*)(o1 + idx);
  const bool k0 = m0[row] >= THRESH;
  const bool k1 = m1[row] >= THRESH;
  *(float4*)(out + idx)                    = k0 ? a : b;
  *(float4*)(out + (size_t)Ntok * C + idx) = k1 ? b : a;
}

// ---------------- host launcher ----------------
extern "C" void kernel_launch(void* const* d_in, const int* /*in_sizes*/,
                              int /*n_in*/, void* d_out, int /*out_size*/,
                              void* d_ws, size_t /*ws_size*/,
                              hipStream_t stream) {
  const float* x[2]    = {(const float*)d_in[0], (const float*)d_in[1]};
  const float* mask[2] = {(const float*)d_in[2], (const float*)d_in[3]};
  const float* Wq    = (const float*)d_in[4];
  const float* Wkv   = (const float*)d_in[5];
  const float* Wsr   = (const float*)d_in[6];
  const float* bsr   = (const float*)d_in[7];
  const float* gamma = (const float*)d_in[8];
  const float* beta  = (const float*)d_in[9];
  const float* Wp    = (const float*)d_in[10];
  const float* bp    = (const float*)d_in[11];

  // bump allocator over workspace, 256B aligned
  char* wp = (char*)d_ws;
  auto alloc = [&](size_t bytes) {
    void* r = wp;
    wp += (bytes + 255) & ~(size_t)255;
    return r;
  };
  const size_t NC = (size_t)Ntok * C;   // 2M elems
  const size_t MC = (size_t)Mred * C;   // 0.5M elems

  bf16* xbf[2]  = {(bf16*)alloc(NC * 2), (bf16*)alloc(NC * 2)};
  bf16* Wq_t    = (bf16*)alloc((size_t)C * C * 2);
  bf16* Wkv_t   = (bf16*)alloc((size_t)2 * C * C * 2);
  bf16* Wsr_t   = (bf16*)alloc((size_t)C * 4 * C * 2);
  bf16* Wp_t    = (bf16*)alloc((size_t)C * C * 2);
  bf16* Qbf[2]  = {(bf16*)alloc(NC * 2), (bf16*)alloc(NC * 2)};
  float* XNf[2] = {(float*)alloc(MC * 4), (float*)alloc(MC * 4)};
  bf16* XNbf[2] = {(bf16*)alloc(MC * 2), (bf16*)alloc(MC * 2)};
  bf16* Kbf[2]  = {(bf16*)alloc(MC * 2), (bf16*)alloc(MC * 2)};
  bf16* Vbf[2]  = {(bf16*)alloc(MC * 2), (bf16*)alloc(MC * 2)};
  bf16* AObf[2] = {(bf16*)alloc(NC * 2), (bf16*)alloc(NC * 2)};
  float* OPf[2] = {(float*)alloc(NC * 4), (float*)alloc(NC * 4)};

  // one-time precision/layout prep
  convert_bf16<<<NC / (256 * 8), 256, 0, stream>>>(x[0], xbf[0]);
  convert_bf16<<<NC / (256 * 8), 256, 0, stream>>>(x[1], xbf[1]);
  transpose_convert<<<dim3(8, 8),  256, 0, stream>>>(Wq,  Wq_t,  C, C);
  transpose_convert<<<dim3(16, 8), 256, 0, stream>>>(Wkv, Wkv_t, C, 2 * C);
  transpose_convert<<<dim3(8, 8),  256, 0, stream>>>(Wp,  Wp_t,  C, C);
  convert_wsr<<<(512 * 2048) / 256, 256, 0, stream>>>(Wsr, Wsr_t);

  for (int b = 0; b < 2; ++b) {
    // Q = (x @ Wq) * 1/8   (softmax scale folded, bf16 out)
    gemm_wmma<0, 1><<<dim3(C / BN, Ntok / BM), 256, 0, stream>>>(
        xbf[b], Wq_t, nullptr, nullptr, nullptr, Qbf[b], nullptr, QSCALE,
        C, C, C, C);
    // xr = SR-conv(x) + bsr  (remapped-K gathered GEMM, f32 out)
    gemm_wmma<1, 0><<<dim3(C / BN, Mred / BM), 256, 0, stream>>>(
        xbf[b], Wsr_t, bsr, nullptr, XNf[b], nullptr, nullptr, 1.f,
        4 * C, 0, 4 * C, C);
    // LayerNorm -> bf16
    layernorm_rows<<<Mred / 8, 256, 0, stream>>>(XNf[b], XNbf[b], gamma, beta,
                                                 Mred);
    // KV = xn @ Wkv -> K row-major bf16, V transposed bf16
    gemm_wmma<0, 2><<<dim3(2 * C / BN, Mred / BM), 256, 0, stream>>>(
        XNbf[b], Wkv_t, nullptr, nullptr, nullptr, Kbf[b], Vbf[b], 1.f,
        C, C, C, 2 * C);
    // attention -> bf16
    attn_wmma<<<dim3(Ntok / BQ, HEADS), 256, 0, stream>>>(Qbf[b], Kbf[b],
                                                          Vbf[b], AObf[b]);
    // out = (attn @ Wp + bp) * mask  (f32 out)
    gemm_wmma<0, 0><<<dim3(C / BN, Ntok / BM), 256, 0, stream>>>(
        AObf[b], Wp_t, bp, mask[b], OPf[b], nullptr, nullptr, 1.f,
        C, C, C, C);
  }
  exchange_kernel<<<(Ntok * C / 4) / 256, 256, 0, stream>>>(
      OPf[0], OPf[1], mask[0], mask[1], (float*)d_out);
}
